// Decoder_24464133718692
// MI455X (gfx1250) — compile-verified, matrix operand order checked by reading, not run on hardware
//
#include <hip/hip_runtime.h>
#include <hip/hip_bf16.h>
#include <math.h>

// Problem constants (from reference setup_inputs)
#define BB   32
#define TT   128
#define EE   256
#define HH   512
#define SS   2048
#define H2   1024   // 2H
#define H3   1536   // 3H
#define KX   1280   // E + 2H   (GRU input width)
#define KP   1792   // E + H + 2H (pre-output width)

typedef __attribute__((ext_vector_type(16))) __bf16 v16bf;
typedef __attribute__((ext_vector_type(8)))  __bf16 v8bf;
typedef __attribute__((ext_vector_type(8)))  float  v8f;
typedef __attribute__((ext_vector_type(4)))  int    v4i;

union FragBF {
    v16bf v;
    v8bf  h[2];
};

// CDNA5 async global->LDS copy path (ASYNCcnt), if the toolchain exposes it.
#if defined(__HIP_DEVICE_COMPILE__) && \
    __has_builtin(__builtin_amdgcn_global_load_async_to_lds_b128)
#define HAVE_ASYNC_LDS 1
#else
#define HAVE_ASYNC_LDS 0
#endif

#if HAVE_ASYNC_LDS
// ISA: dsaddr = VDST + INST_OFFSET and memaddr = VADDR + INST_OFFSET, so one
// base pair + literal offsets copies a contiguous run.  The builtin takes
// 128-bit (v4i) pointers in AS1 (global) / AS3 (LDS).
#define ASYNC_CP16(g, l, off)                                             \
    __builtin_amdgcn_global_load_async_to_lds_b128(                       \
        (__attribute__((address_space(1))) v4i*)(g),                      \
        (__attribute__((address_space(3))) v4i*)(l), (off), 0)
#endif

// ---------------------------------------------------------------------------
// f32 -> bf16 cast (grid-stride)
// ---------------------------------------------------------------------------
__global__ void k_cast_f32_bf16(const float* __restrict__ src,
                                __bf16* __restrict__ dst, size_t n) {
    size_t i = (size_t)blockIdx.x * blockDim.x + threadIdx.x;
    size_t stride = (size_t)gridDim.x * blockDim.x;
    for (; i < n; i += stride) dst[i] = (__bf16)src[i];
}

// ---------------------------------------------------------------------------
// Bridge: h0 = tanh(encoder_final @ Wbridge.T + bbridge)    (B x H)
// ---------------------------------------------------------------------------
__global__ void k_bridge(const float* __restrict__ enc_final,   // (B, 2H)
                         const float* __restrict__ Wbridge,     // (H, 2H)
                         const float* __restrict__ bbridge,     // (H)
                         float* __restrict__ h,                 // (B, H)
                         __bf16* __restrict__ h_b16) {          // (B, H)
    int tid = blockIdx.x * blockDim.x + threadIdx.x;
    if (tid >= BB * HH) return;
    int b = tid / HH, n = tid % HH;
    const float* xr = enc_final + (size_t)b * H2;
    const float* wr = Wbridge + (size_t)n * H2;
    float acc = bbridge[n];
    for (int k = 0; k < H2; ++k) acc += xr[k] * wr[k];
    float v = tanhf(acc);
    h[tid] = v;
    h_b16[tid] = (__bf16)v;
}

// ---------------------------------------------------------------------------
// WMMA bf16 GEMM:  C[m,n] = sum_k A[m,k] * Bw[n,k]  (+ bias[n])
// A: (M,K) bf16 row-major, lda;  Bw: (N,K) bf16 row-major, ldb
//
// Block = 64 threads (2 waves).  Block tile = 32 (M) x 64 (N).
//   - wave w handles M-subtile w (16 rows), N-blocked over 4 tiles:
//     one A fragment feeds 4 WMMAs (4x cut in A traffic).
//   - B panel (64 rows x 64 K) staged in LDS per K-chunk, shared by both
//     waves; staged via GLOBAL_LOAD_ASYNC_TO_LDS_B128 when available.
// K and N must be multiples of 64, M a multiple of 32 (all true here).
//
// Fragment layout per CDNA5 ISA 7.12.2 (16-bit A 16x32 / B 32x16):
//   lane row/col = tile*16 + (lane&15); kg = (lane>>4)*8
//   halves j=0..7  -> K = k0 + kg + j ; j=8..15 -> K = k0 + 16 + kg + (j-8)
//   C/D: row = (lane>=16 ? 8:0)+r, col = lane&15
// ---------------------------------------------------------------------------
#define NT_N 4
#define BPAD 72   // 64 + 8 bf16 pad -> 144B row stride, conflict-free ds_load

__global__ __launch_bounds__(64)
void k_gemm_wmma_bt(const __bf16* __restrict__ A,
                    const __bf16* __restrict__ Bw,
                    const float* __restrict__ bias,
                    void* __restrict__ Cout, int out_bf16,
                    int K, int lda, int ldb, int ldc) {
    __shared__ __bf16 bsh[64][BPAD];

    const int tid   = threadIdx.x;
    const int lane  = tid & 31;
    const int wave  = tid >> 5;                 // 0..1 -> M subtile
    const int n0    = blockIdx.x * 64;
    const int m0    = blockIdx.y * 32 + wave * 16;
    const int rlane = lane & 15;
    const int kg    = (lane >> 4) * 8;

    const __bf16* ap   = A  + (size_t)(m0 + rlane) * lda + kg;
    const __bf16* brow = Bw + (size_t)(n0 + tid) * ldb;  // row staged by tid

    v8f acc[NT_N] = {};

    for (int k0 = 0; k0 < K; k0 += 64) {
        __syncthreads();   // previous chunk's LDS reads complete
        // ---- stage B panel rows n0..n0+63, cols k0..k0+63 (128B/thread) ----
        {
            const __bf16* g = brow + k0;
            __bf16* l = &bsh[tid][0];
#if HAVE_ASYNC_LDS
            ASYNC_CP16(g, l, 0);
            ASYNC_CP16(g, l, 16);
            ASYNC_CP16(g, l, 32);
            ASYNC_CP16(g, l, 48);
            ASYNC_CP16(g, l, 64);
            ASYNC_CP16(g, l, 80);
            ASYNC_CP16(g, l, 96);
            ASYNC_CP16(g, l, 112);
#if __has_builtin(__builtin_amdgcn_s_wait_asynccnt)
            __builtin_amdgcn_s_wait_asynccnt(0);
#else
            asm volatile("s_wait_asynccnt 0" ::: "memory");
#endif
#else
            const v8bf* gv = (const v8bf*)g;
            v8bf* lv = (v8bf*)l;
#pragma unroll
            for (int i = 0; i < 8; ++i) lv[i] = gv[i];
#endif
        }
        __syncthreads();

        // ---- compute: 2 K-substeps of 32, 4 N-tiles each ----
#pragma unroll
        for (int ks = 0; ks < 2; ++ks) {
            FragBF a;
            const __bf16* ap2 = ap + k0 + ks * 32;
            a.h[0] = *(const v8bf*)(ap2);
            a.h[1] = *(const v8bf*)(ap2 + 16);
#pragma unroll
            for (int nt = 0; nt < NT_N; ++nt) {
                FragBF b;
                const __bf16* bp = &bsh[nt * 16 + rlane][ks * 32 + kg];
                b.h[0] = *(const v8bf*)(bp);
                b.h[1] = *(const v8bf*)(bp + 16);
                acc[nt] = __builtin_amdgcn_wmma_f32_16x16x32_bf16(
                    /*neg_a=*/false, a.v, /*neg_b=*/false, b.v,
                    /*c_mod=*/(short)0, acc[nt],
                    /*reuse_a=*/false, /*reuse_b=*/false);
            }
        }
    }

    const int rbase = m0 + ((lane >> 4) * 8);
#pragma unroll
    for (int nt = 0; nt < NT_N; ++nt) {
        const int col = n0 + nt * 16 + rlane;
        const float bv = bias ? bias[col] : 0.0f;
        if (out_bf16) {
            __bf16* C = (__bf16*)Cout;
#pragma unroll
            for (int r = 0; r < 8; ++r)
                C[(size_t)(rbase + r) * ldc + col] = (__bf16)(acc[nt][r] + bv);
        } else {
            float* C = (float*)Cout;
#pragma unroll
            for (int r = 0; r < 8; ++r)
                C[(size_t)(rbase + r) * ldc + col] = acc[nt][r] + bv;
        }
    }
}

// ---------------------------------------------------------------------------
// Attention step: one block per batch row b.
//   q = h_b @ Wquery.T                         (H)
//   scores[s] = sum_h tanh(q[h]+pk[b,s,h])*e[h]; masked softmax over S
//   context[b,:] = sum_s alpha[s] * enc[b,s,:]  (2H)
// pk and enc are bf16 (L2-resident working set: 64MB + 128MB = 192MB = L2).
// ---------------------------------------------------------------------------
__global__ __launch_bounds__(256)
void k_attention(const float* __restrict__ h,        // (B,H)
                 const float* __restrict__ Wquery,   // (H,H)
                 const float* __restrict__ Wenergy,  // (1,H)
                 const __bf16* __restrict__ pk,      // (B,S,H)
                 const __bf16* __restrict__ encb,    // (B,S,2H)
                 const unsigned char* __restrict__ mask, // (B,1,S)
                 float* __restrict__ context) {      // (B,2H)
    const int b   = blockIdx.x;
    const int tid = threadIdx.x;
    const int NT  = 256;

    __shared__ float hs[HH];
    __shared__ float es[HH];
    __shared__ float qs[HH];
    __shared__ float sc[SS];
    __shared__ float red[256];

    for (int k = tid; k < HH; k += NT) {
        hs[k] = h[(size_t)b * HH + k];
        es[k] = Wenergy[k];
    }
    __syncthreads();

    // q = h @ Wquery.T
    for (int i = tid; i < HH; i += NT) {
        const float* wr = Wquery + (size_t)i * HH;
        float acc = 0.f;
        for (int k = 0; k < HH; ++k) acc += hs[k] * wr[k];
        qs[i] = acc;
    }
    __syncthreads();

    // scores
    for (int s = tid; s < SS; s += NT) {
        const v8bf* p = (const v8bf*)(pk + ((size_t)b * SS + s) * HH);
        __builtin_prefetch(pk + ((size_t)b * SS + s + NT) * HH, 0, 1);
        float acc = 0.f;
        for (int h8 = 0; h8 < HH / 8; ++h8) {
            v8bf v = p[h8];
#pragma unroll
            for (int j = 0; j < 8; ++j)
                acc += tanhf(qs[h8 * 8 + j] + (float)v[j]) * es[h8 * 8 + j];
        }
        sc[s] = mask[(size_t)b * SS + s] ? acc : -1.0e9f;
    }
    __syncthreads();

    // softmax: max
    float lm = -INFINITY;
    for (int s = tid; s < SS; s += NT) lm = fmaxf(lm, sc[s]);
    red[tid] = lm; __syncthreads();
    for (int off = 128; off > 0; off >>= 1) {
        if (tid < off) red[tid] = fmaxf(red[tid], red[tid + off]);
        __syncthreads();
    }
    const float mx = red[0]; __syncthreads();

    // exp + sum
    float ls = 0.f;
    for (int s = tid; s < SS; s += NT) {
        float v = __expf(sc[s] - mx);
        sc[s] = v;
        ls += v;
    }
    red[tid] = ls; __syncthreads();
    for (int off = 128; off > 0; off >>= 1) {
        if (tid < off) red[tid] += red[tid + off];
        __syncthreads();
    }
    const float inv = 1.0f / red[0]; __syncthreads();
    for (int s = tid; s < SS; s += NT) sc[s] *= inv;
    __syncthreads();

    // context = alphas @ enc   (coalesced across k)
    for (int k = tid; k < H2; k += NT) {
        const __bf16* ep = encb + (size_t)b * SS * H2 + k;
        float acc = 0.f;
        for (int s = 0; s < SS; ++s) acc += sc[s] * (float)ep[(size_t)s * H2];
        context[(size_t)b * H2 + k] = acc;
    }
}

// ---------------------------------------------------------------------------
// Assemble x = [x_t, context] as bf16     (B, KX)
// ---------------------------------------------------------------------------
__global__ void k_xcat(const float* __restrict__ trg_embed,  // (B,T,E)
                       const float* __restrict__ context,    // (B,2H)
                       __bf16* __restrict__ xcat, int t) {
    int tid = blockIdx.x * blockDim.x + threadIdx.x;
    if (tid >= BB * KX) return;
    int b = tid / KX, j = tid % KX;
    float v = (j < EE) ? trg_embed[((size_t)b * TT + t) * EE + j]
                       : context[(size_t)b * H2 + (j - EE)];
    xcat[tid] = (__bf16)v;
}

// ---------------------------------------------------------------------------
// GRU elementwise: gates -> h_new; writes h, h_b16, decoder_states, hT
// ---------------------------------------------------------------------------
__global__ void k_gru(const float* __restrict__ gi,   // (B,3H)
                      const float* __restrict__ gh,   // (B,3H)
                      float* __restrict__ h,          // (B,H) in/out
                      __bf16* __restrict__ h_b16,     // (B,H)
                      float* __restrict__ dec_out,    // (B,T,H)
                      float* __restrict__ hT_out,     // (B,H)
                      int t) {
    int tid = blockIdx.x * blockDim.x + threadIdx.x;
    if (tid >= BB * HH) return;
    int b = tid / HH, n = tid % HH;
    size_t gb = (size_t)b * H3;
    float ir = gi[gb + n], iz = gi[gb + HH + n], in_ = gi[gb + H2 + n];
    float hr = gh[gb + n], hz = gh[gb + HH + n], hn  = gh[gb + H2 + n];
    float r = 1.0f / (1.0f + __expf(-(ir + hr)));
    float z = 1.0f / (1.0f + __expf(-(iz + hz)));
    float nn = tanhf(in_ + r * hn);
    float hp = h[tid];
    float hv = (1.0f - z) * nn + z * hp;
    h[tid] = hv;
    h_b16[tid] = (__bf16)hv;
    dec_out[((size_t)b * TT + t) * HH + n] = hv;
    hT_out[(size_t)b * HH + n] = hv;   // last step's write survives
}

// ---------------------------------------------------------------------------
// Assemble pre-cat = [x_t, h_new, context] as bf16   (B, KP)
// ---------------------------------------------------------------------------
__global__ void k_precat(const float* __restrict__ trg_embed,
                         const float* __restrict__ h,
                         const float* __restrict__ context,
                         __bf16* __restrict__ precat, int t) {
    int tid = blockIdx.x * blockDim.x + threadIdx.x;
    if (tid >= BB * KP) return;
    int b = tid / KP, j = tid % KP;
    float v;
    if (j < EE)            v = trg_embed[((size_t)b * TT + t) * EE + j];
    else if (j < EE + HH)  v = h[(size_t)b * HH + (j - EE)];
    else                   v = context[(size_t)b * H2 + (j - EE - HH)];
    precat[tid] = (__bf16)v;
}

// ---------------------------------------------------------------------------
// Host launcher
// ---------------------------------------------------------------------------
extern "C" void kernel_launch(void* const* d_in, const int* in_sizes, int n_in,
                              void* d_out, int out_size, void* d_ws, size_t ws_size,
                              hipStream_t stream) {
    const float* trg_embed  = (const float*)d_in[0];   // (B,T,E)
    const float* enc_hidden = (const float*)d_in[1];   // (B,S,2H)
    const float* enc_final  = (const float*)d_in[2];   // (B,2H)
    const unsigned char* src_mask = (const unsigned char*)d_in[3]; // (B,1,S) bool
    const float* Wkey    = (const float*)d_in[4];      // (H,2H)
    const float* Wquery  = (const float*)d_in[5];      // (H,H)
    const float* Wenergy = (const float*)d_in[6];      // (1,H)
    const float* W_ih    = (const float*)d_in[7];      // (3H, E+2H)
    const float* W_hh    = (const float*)d_in[8];      // (3H, H)
    const float* b_ih    = (const float*)d_in[9];      // (3H)
    const float* b_hh    = (const float*)d_in[10];     // (3H)
    const float* Wbridge = (const float*)d_in[11];     // (H,2H)
    const float* bbridge = (const float*)d_in[12];     // (H)
    const float* Wpre    = (const float*)d_in[13];     // (H, KP)

    float* out = (float*)d_out;
    float* dec = out;                                   // (B,T,H)
    float* hT  = out + (size_t)BB * TT * HH;            // (B,H)
    float* pre = hT + (size_t)BB * HH;                  // (B,T,H)

    // workspace layout
    char* base = (char*)d_ws;
    size_t off = 0;
    auto alloc = [&](size_t bytes) -> char* {
        char* p = base + off;
        off = (off + bytes + 255) & ~(size_t)255;
        return p;
    };
    __bf16* enc_b  = (__bf16*)alloc((size_t)BB * SS * H2 * 2);  // 128 MB
    __bf16* pk_b   = (__bf16*)alloc((size_t)BB * SS * HH * 2);  //  64 MB
    __bf16* wkey_b = (__bf16*)alloc((size_t)HH * H2 * 2);
    __bf16* wih_b  = (__bf16*)alloc((size_t)H3 * KX * 2);
    __bf16* whh_b  = (__bf16*)alloc((size_t)H3 * HH * 2);
    __bf16* wpre_b = (__bf16*)alloc((size_t)HH * KP * 2);
    float*  hbuf   = (float*)alloc((size_t)BB * HH * 4);
    __bf16* h_b16  = (__bf16*)alloc((size_t)BB * HH * 2);
    float*  ctx    = (float*)alloc((size_t)BB * H2 * 4);
    __bf16* xcat   = (__bf16*)alloc((size_t)BB * KX * 2);
    __bf16* precat = (__bf16*)alloc((size_t)BB * KP * 2);
    float*  gi     = (float*)alloc((size_t)BB * H3 * 4);
    float*  gh     = (float*)alloc((size_t)BB * H3 * 4);
    (void)ws_size; (void)n_in; (void)in_sizes; (void)out_size;

    // one-time casts to bf16 (recomputed every call: deterministic)
    k_cast_f32_bf16<<<2048, 256, 0, stream>>>(enc_hidden, enc_b, (size_t)BB * SS * H2);
    k_cast_f32_bf16<<<512, 256, 0, stream>>>(Wkey,  wkey_b, (size_t)HH * H2);
    k_cast_f32_bf16<<<512, 256, 0, stream>>>(W_ih,  wih_b,  (size_t)H3 * KX);
    k_cast_f32_bf16<<<512, 256, 0, stream>>>(W_hh,  whh_b,  (size_t)H3 * HH);
    k_cast_f32_bf16<<<512, 256, 0, stream>>>(Wpre,  wpre_b, (size_t)HH * KP);

    // bridge: h0
    k_bridge<<<(BB * HH + 255) / 256, 256, 0, stream>>>(enc_final, Wbridge, bbridge,
                                                        hbuf, h_b16);

    // proj_key = enc @ Wkey.T  -> bf16 (B*S, H); M=65536, N=512, K=1024
    {
        dim3 grid(HH / 64, (BB * SS) / 32);
        k_gemm_wmma_bt<<<grid, 64, 0, stream>>>(enc_b, wkey_b, nullptr,
                                                (void*)pk_b, /*bf16*/1,
                                                /*K=*/H2, /*lda=*/H2, /*ldb=*/H2,
                                                /*ldc=*/HH);
    }

    // sequential scan
    for (int t = 0; t < TT; ++t) {
        k_attention<<<BB, 256, 0, stream>>>(hbuf, Wquery, Wenergy, pk_b, enc_b,
                                            src_mask, ctx);

        k_xcat<<<(BB * KX + 255) / 256, 256, 0, stream>>>(trg_embed, ctx, xcat, t);

        // gi = xcat @ W_ih.T + b_ih : M=32, N=1536, K=1280
        {
            dim3 grid(H3 / 64, BB / 32);
            k_gemm_wmma_bt<<<grid, 64, 0, stream>>>(xcat, wih_b, b_ih,
                                                    (void*)gi, 0,
                                                    KX, KX, KX, H3);
        }
        // gh = h @ W_hh.T + b_hh : M=32, N=1536, K=512
        {
            dim3 grid(H3 / 64, BB / 32);
            k_gemm_wmma_bt<<<grid, 64, 0, stream>>>(h_b16, whh_b, b_hh,
                                                    (void*)gh, 0,
                                                    HH, HH, HH, H3);
        }

        k_gru<<<(BB * HH + 255) / 256, 256, 0, stream>>>(gi, gh, hbuf, h_b16,
                                                         dec, hT, t);

        k_precat<<<(BB * KP + 255) / 256, 256, 0, stream>>>(trg_embed, hbuf, ctx,
                                                            precat, t);

        // pre[:, t, :] = precat @ Wpre.T : M=32, N=512, K=1792, ldc = T*H
        {
            dim3 grid(HH / 64, BB / 32);
            k_gemm_wmma_bt<<<grid, 64, 0, stream>>>(precat, wpre_b, nullptr,
                                                    (void*)(pre + (size_t)t * HH), 0,
                                                    KP, KP, KP, TT * HH);
        }
    }
}
